// GGNN_52759378264071
// MI455X (gfx1250) — matrix operand order repaired and checked by reading, not compiled, not run on hardware
//
#include <hip/hip_runtime.h>
#include <hip/hip_bf16.h>
#include <math.h>

// Problem constants (B, N, D, A, E, T, H, O) = (2, 2048, 64, 16, 2, 5, 128, 1)
enum : long {
  Bb = 2, Nn = 2048, Dd = 64, Aa = 16, Ee = 2, Tt = 5, Hh = 128,
  NE2  = 2 * Nn * Ee,          // 8192 adj row length
  KAGG = Nn * Ee,              // 4096 aggregation K
  HN   = Bb * Nn * Dd,         // 262144 elems of h
  INS1 = Bb * Dd * Nn * Ee,    // 524288 elems per direction of msgs^T
};

typedef __attribute__((ext_vector_type(16))) __bf16 v16bf;
typedef __attribute__((ext_vector_type(8)))  __bf16 v8bf;
typedef __attribute__((ext_vector_type(8)))  float  v8f;

// ---------------------------------------------------------------- conversions
__global__ void f32_to_bf16_kernel(const float* __restrict__ in,
                                   __bf16* __restrict__ out, long n) {
  long i = (long)blockIdx.x * blockDim.x + threadIdx.x;
  long s = (long)gridDim.x * blockDim.x;
  for (; i < n; i += s) out[i] = (__bf16)in[i];
}

// W[e][d][k] (f32) -> Wt[e][k][d] (bf16)  : column-major B operand for WMMA
__global__ void w_transpose_bf16(const float* __restrict__ W,
                                 __bf16* __restrict__ Wt, int n) {
  int i = blockIdx.x * blockDim.x + threadIdx.x;
  if (i < n) {
    int e = i / (64 * 64), r = i % (64 * 64), d = r / 64, k = r % 64;
    Wt[e * 4096 + k * 64 + d] = (__bf16)W[i];
  }
}

// ---------------------------------------------------------------- WMMA GEMM
// One wave computes TWO adjacent 16x16 f32 tiles of C = A(bf16,MxK) @ B,
// sharing the A fragment.  B is stored COLUMN-MAJOR (ldB = column stride) so
// every lane's 16 K-values are two contiguous b128 loads.
// Optional per-column f32 bias; stores f32 row-major and/or bf16 TRANSPOSED
// (column-major, fragment-friendly for a later GEMM's B operand) via one
// packed v8bf store per lane per tile.
// blockIdx.z = b*nE + e selects batch/edge-type slices via flat strides.
#define WPB 4  // waves (16-row tiles) per block
template <bool HAS_BIAS, bool STORE_F32, bool STORE_BF16T>
__global__ __launch_bounds__(32 * WPB)
void wmma_gemm_bf16(const __bf16* __restrict__ A,  long sAb, long sAe, int ldA,
                    const __bf16* __restrict__ Bt, long sBb, long sBe, int ldB,
                    float* __restrict__ Cf, long sCfb, long sCfe, int ldCf,
                    __bf16* __restrict__ Ct, long sCtb, long sCte, int ldCt,
                    const float* __restrict__ bias, int sBiasE,
                    int K, int nE) {
  const int z    = blockIdx.z;
  const int b    = z / nE;
  const int e    = z % nE;
  const int lane = threadIdx.x & 31;
  const int w    = threadIdx.x >> 5;
  const int tM   = blockIdx.x * WPB + w;
  const int lo16 = lane & 15;
  const int hi   = lane >> 4;

  const int rowA = tM * 16 + lo16;                 // A row this lane supplies
  const int col0 = blockIdx.y * 32 + lo16;         // C column, tile 0
  const int col1 = col0 + 16;                      // C column, tile 1

  const __bf16* Ap  = A  + sAb * b + sAe * e + (long)rowA * ldA + hi * 16;
  const __bf16* Bp0 = Bt + sBb * b + sBe * e + (long)col0 * ldB;
  const __bf16* Bp1 = Bp0 + (long)16 * ldB;

  v8f acc0, acc1;
  const float bi0 = HAS_BIAS ? bias[e * sBiasE + col0] : 0.0f;
  const float bi1 = HAS_BIAS ? bias[e * sBiasE + col1] : 0.0f;
#pragma unroll
  for (int i = 0; i < 8; ++i) { acc0[i] = bi0; acc1[i] = bi1; }

  for (int k0 = 0; k0 < K; k0 += 32) {
    // A fragment: elem e2 -> K = k0 + 16*hi + e2
    v8bf alo = *(const v8bf*)(Ap + k0);
    v8bf ahi = *(const v8bf*)(Ap + k0 + 8);
    v16bf afrag = __builtin_shufflevector(alo, ahi,
        0, 1, 2, 3, 4, 5, 6, 7, 8, 9, 10, 11, 12, 13, 14, 15);
    __builtin_prefetch(Ap + k0 + 128, 0, 1);       // global_prefetch ahead

    // B fragments (column-major): elem e2 -> K = k0 + 16*hi + e2
    const int kb = k0 + hi * 16;
    v8bf b0lo = *(const v8bf*)(Bp0 + kb);
    v8bf b0hi = *(const v8bf*)(Bp0 + kb + 8);
    v8bf b1lo = *(const v8bf*)(Bp1 + kb);
    v8bf b1hi = *(const v8bf*)(Bp1 + kb + 8);
    v16bf bf0 = __builtin_shufflevector(b0lo, b0hi,
        0, 1, 2, 3, 4, 5, 6, 7, 8, 9, 10, 11, 12, 13, 14, 15);
    v16bf bf1 = __builtin_shufflevector(b1lo, b1hi,
        0, 1, 2, 3, 4, 5, 6, 7, 8, 9, 10, 11, 12, 13, 14, 15);

    acc0 = __builtin_amdgcn_wmma_f32_16x16x32_bf16(
        false, afrag, false, bf0, (short)0, acc0, false, false);
    acc1 = __builtin_amdgcn_wmma_f32_16x16x32_bf16(
        false, afrag, false, bf1, (short)0, acc1, false, false);
  }

  const int mbase = tM * 16 + hi * 8;              // this lane's 8 rows start
  if (STORE_F32) {
    const long cb = sCfb * b + sCfe * e;
#pragma unroll
    for (int i = 0; i < 8; ++i) {
      Cf[cb + (long)(mbase + i) * ldCf + col0] = acc0[i];
      Cf[cb + (long)(mbase + i) * ldCf + col1] = acc1[i];
    }
  }
  if (STORE_BF16T) {
    // transposed store: Ct[col * ldCt + m], 8 contiguous m per lane -> b128
    v8bf p0, p1;
#pragma unroll
    for (int i = 0; i < 8; ++i) { p0[i] = (__bf16)acc0[i]; p1[i] = (__bf16)acc1[i]; }
    const long cb = sCtb * b + sCte * e + mbase;
    *(v8bf*)(Ct + cb + (long)col0 * ldCt) = p0;
    *(v8bf*)(Ct + cb + (long)col1 * ldCt) = p1;
  }
}

// ---------------------------------------------------------------- fused GRU
// One 64-thread group per node row: a = [a_in, a_out, h] (192),
// r = sig(a@Wr+br), z = sig(a@Wz+bz), h_hat = tanh([a_in,a_out,r*h]@Wh+bh),
// h' = (1-z)h + z*h_hat.  4 rows per 256-thread block; a staged in LDS.
__global__ void gru_fused(const float* __restrict__ ain,  const float* __restrict__ aout,
                          const float* __restrict__ h,    float* __restrict__ hn,
                          const float* __restrict__ Wr,   const float* __restrict__ br,
                          const float* __restrict__ Wz,   const float* __restrict__ bz,
                          const float* __restrict__ Wh,   const float* __restrict__ bh) {
  __shared__ float as[4][192];
  const int  r = threadIdx.x >> 6;
  const int  d = threadIdx.x & 63;
  const long g = (long)blockIdx.x * 4 + r;        // node row in [0, B*N)

  const float hd = h[g * 64 + d];
  as[r][d]       = ain [g * 64 + d];
  as[r][64 + d]  = aout[g * 64 + d];
  as[r][128 + d] = hd;
  __syncthreads();

  float sr = br[d], sz = bz[d];
#pragma unroll 8
  for (int k = 0; k < 192; ++k) {
    const float av = as[r][k];
    sr += av * Wr[k * 64 + d];
    sz += av * Wz[k * 64 + d];
  }
  const float rr = 1.0f / (1.0f + __expf(-sr));
  const float zz = 1.0f / (1.0f + __expf(-sz));
  __syncthreads();
  as[r][128 + d] = rr * hd;                       // joint's last third = r*h
  __syncthreads();

  float sh = bh[d];
#pragma unroll 8
  for (int k = 0; k < 192; ++k) sh += as[r][k] * Wh[k * 64 + d];
  const float hh = tanhf(sh);
  hn[g * 64 + d] = (1.0f - zz) * hd + zz * hh;
}

// ---------------------------------------------------------------- readout
__global__ void readout(const float* __restrict__ h,   const float* __restrict__ ann,
                        const float* __restrict__ Wo1, const float* __restrict__ bo1,
                        const float* __restrict__ Wo2, const float* __restrict__ bo2,
                        float* __restrict__ out) {
  __shared__ float xs[80];
  __shared__ float red[128];
  const long g = blockIdx.x;
  const int  j = threadIdx.x;
  if (j < 64)      xs[j] = h[g * 64 + j];
  else if (j < 80) xs[j] = ann[g * 16 + (j - 64)];
  __syncthreads();

  float s = bo1[j];
#pragma unroll 8
  for (int k = 0; k < 80; ++k) s += xs[k] * Wo1[k * 128 + j];
  const float y = 1.0f / (1.0f + __expf(-s));
  red[j] = y * Wo2[j];
  __syncthreads();
#pragma unroll
  for (int off = 64; off > 0; off >>= 1) {
    if (j < off) red[j] += red[j + off];
    __syncthreads();
  }
  if (j == 0) out[g] = red[0] + bo2[0];
}

// ---------------------------------------------------------------- launcher
extern "C" void kernel_launch(void* const* d_in, const int* in_sizes, int n_in,
                              void* d_out, int out_size, void* d_ws, size_t ws_size,
                              hipStream_t stream) {
  const float* prop  = (const float*)d_in[0];
  const float* ann   = (const float*)d_in[1];
  const float* adj   = (const float*)d_in[2];
  const float* W_in  = (const float*)d_in[3];
  const float* b_in  = (const float*)d_in[4];
  const float* W_out = (const float*)d_in[5];
  const float* b_out = (const float*)d_in[6];
  const float* Wr    = (const float*)d_in[7];
  const float* br    = (const float*)d_in[8];
  const float* Wz    = (const float*)d_in[9];
  const float* bz    = (const float*)d_in[10];
  const float* Wh    = (const float*)d_in[11];
  const float* bh    = (const float*)d_in[12];
  const float* Wo1   = (const float*)d_in[13];
  const float* bo1   = (const float*)d_in[14];
  const float* Wo2   = (const float*)d_in[15];
  const float* bo2   = (const float*)d_in[16];
  float* out = (float*)d_out;

  // Workspace layout (bytes; every section is 256-aligned by construction)
  char* ws = (char*)d_ws;
  const long adj_bytes = Bb * Nn * NE2 * 2;                 // 67,108,864
  __bf16* adjb   = (__bf16*)(ws);
  __bf16* wbfT   = (__bf16*)(ws + adj_bytes);                              // WinT|WoutT bf16 (32KB)
  __bf16* hbf    = (__bf16*)(ws + adj_bytes + 32768);                      // h bf16 (512KB)
  __bf16* msgsT  = (__bf16*)(ws + adj_bytes + 32768 + 524288);             // (dir,b,d,e*N+n) bf16 (2MB)
  float*  ainout = (float*) (ws + adj_bytes + 32768 + 524288 + 2097152);   // (dir,b,n,d) f32 (2MB)
  float*  h0     = (float*) (ws + adj_bytes + 32768 + 524288 + 2097152 + 2097152);
  float*  h1     = h0 + HN;

  // h <- prop_state (inputs must not be mutated)
  hipMemcpyAsync(h0, prop, HN * sizeof(float), hipMemcpyDeviceToDevice, stream);

  // One-shot conversions: adjacency to bf16 (halves re-read bytes; ~67MB sits
  // in the 192MB L2 across all 5 steps) + transposed bf16 weights.
  f32_to_bf16_kernel<<<2048, 256, 0, stream>>>(adj, adjb, Bb * Nn * NE2);
  w_transpose_bf16<<<32, 256, 0, stream>>>(W_in,  wbfT,        (int)(Ee * Dd * Dd));
  w_transpose_bf16<<<32, 256, 0, stream>>>(W_out, wbfT + 8192, (int)(Ee * Dd * Dd));

  float* hc = h0;
  float* hn = h1;
  const dim3 gemm_grid(Nn / 16 / WPB, Dd / 32, 4);   // (32, 2, 4)

  for (int t = 0; t < Tt; ++t) {
    f32_to_bf16_kernel<<<256, 256, 0, stream>>>(hc, hbf, HN);

    // msgsT[dir=0][b][:, e*N+n] = (h[b] @ W_in[e] + b_in[e])^T   (bf16, col-major)
    wmma_gemm_bf16<true, false, true><<<gemm_grid, 32 * WPB, 0, stream>>>(
        hbf, (long)Nn * Dd, 0L, (int)Dd,
        wbfT, 0L, (long)Dd * Dd, (int)Dd,
        nullptr, 0L, 0L, 0,
        msgsT, (long)Dd * KAGG, (long)Nn, (int)KAGG,
        b_in, (int)Dd, (int)Dd, (int)Ee);
    // msgsT[dir=1] from W_out / b_out
    wmma_gemm_bf16<true, false, true><<<gemm_grid, 32 * WPB, 0, stream>>>(
        hbf, (long)Nn * Dd, 0L, (int)Dd,
        wbfT + 8192, 0L, (long)Dd * Dd, (int)Dd,
        nullptr, 0L, 0L, 0,
        msgsT + INS1, (long)Dd * KAGG, (long)Nn, (int)KAGG,
        b_out, (int)Dd, (int)Dd, (int)Ee);

    // a_in/a_out = adj_slice(N x 4096, bf16) @ msgs(4096 x 64)  -> f32 row-major
    wmma_gemm_bf16<false, true, false><<<gemm_grid, 32 * WPB, 0, stream>>>(
        adjb, (long)Nn * NE2, (long)KAGG, (int)NE2,
        msgsT, (long)Dd * KAGG, (long)INS1, (int)KAGG,
        ainout, (long)Nn * Dd, (long)Bb * Nn * Dd, (int)Dd,
        nullptr, 0L, 0L, 0,
        nullptr, 0, (int)KAGG, 2);

    // fused GRU cell
    gru_fused<<<(Bb * Nn) / 4, 256, 0, stream>>>(
        ainout, ainout + Bb * Nn * Dd, hc, hn, Wr, br, Wz, bz, Wh, bh);
    float* tmp = hc; hc = hn; hn = tmp;
  }

  readout<<<Bb * Nn, 128, 0, stream>>>(hc, ann, Wo1, bo1, Wo2, bo2, out);
}
